// Attention_41515153883473
// MI455X (gfx1250) — matrix operand order repaired and checked by reading, not compiled, not run on hardware
//
#include <hip/hip_runtime.h>

// ---------------- problem constants ----------------
#define BATCH 2
#define SEQ   2048
#define DEM   1024      // d_model
#define NH    16        // heads
#define DKK   64        // d_k
#define N3    3072      // 3*d_model
#define NEG_BIG (-9.2233720368547758e18f)   // -2^63 as float

typedef __attribute__((ext_vector_type(16))) __bf16 v16bf;
typedef __attribute__((ext_vector_type(8)))  __bf16 v8bf;
typedef __attribute__((ext_vector_type(8)))  float  v8f;

static __device__ __forceinline__ __bf16 f2bf(float f) {
  union { float f; unsigned u; } x; x.f = f;
  unsigned r = x.u + 0x7FFFu + ((x.u >> 16) & 1u);   // round-to-nearest-even
  union { unsigned short s; __bf16 b; } y; y.s = (unsigned short)(r >> 16);
  return y.b;
}

// ---------------------------------------------------------------------------
// 1) convert x -> bf16 row-major; wk_w, wo_w -> bf16 TRANSPOSED (out-channel
//    major) so WMMA B-fragments are contiguous 32B loads along K.
// ---------------------------------------------------------------------------
__global__ __launch_bounds__(256) void k_convert(
    const float* __restrict__ x, const float* __restrict__ wk,
    const float* __restrict__ wo,
    __bf16* __restrict__ xb, __bf16* __restrict__ wkT, __bf16* __restrict__ woT) {
  int i = blockIdx.x * 256 + threadIdx.x;
  if (i < BATCH * SEQ * DEM) xb[i] = f2bf(x[i]);
  if (i < N3 * DEM) {                       // wkT[n*1024 + k] = wk[k*3072 + n]
    int n = i / DEM, k = i % DEM;
    wkT[i] = f2bf(wk[(size_t)k * N3 + n]);
  }
  if (i < DEM * DEM) {                      // woT[n*1024 + k] = wo[k*1024 + n]
    int n = i / DEM, k = i % DEM;
    woT[i] = f2bf(wo[(size_t)k * DEM + n]);
  }
}

// ---------------------------------------------------------------------------
// 2) QKV GEMM: [4096 x 1024] x [1024 x 3072] (+bias).
//    Block = 8 waves, tile 64 rows x 256 cols; wave = 32x64 (2 m-tiles x
//    4 n-tiles = 8 WMMAs per K-step; B fragments reused across m-tiles).
//    Q scaled by 1/sqrt(64)=0.125, stored [b,h,s,dk]; K stored [b,h,s,dk];
//    V stored TRANSPOSED [b,h,dk,s] for the attn@V B-operand.
// ---------------------------------------------------------------------------
__global__ __launch_bounds__(256) void k_qkv(
    const __bf16* __restrict__ xb, const __bf16* __restrict__ wkT,
    const float* __restrict__ bias,
    __bf16* __restrict__ qb, __bf16* __restrict__ kbuf, __bf16* __restrict__ vT) {
  const int w = threadIdx.x >> 5, lane = threadIdx.x & 31;
  const int hi = lane >> 4, ln = lane & 15;
  const int mBase = blockIdx.x * 64 + (w & 1) * 32;     // wave: 32 rows
  const int nBase = blockIdx.y * 256 + (w >> 1) * 64;   // wave: 64 cols

  v8f acc[2][4] = {};
  const __bf16* arow0 = xb + (size_t)(mBase + ln) * DEM;       // m-tile 0, row M=ln
  const __bf16* arow1 = xb + (size_t)(mBase + 16 + ln) * DEM;  // m-tile 1
  for (int ks = 0; ks < DEM; ks += 32) {
    v16bf A0, A1;
    v8bf a00 = *(const v8bf*)(arow0 + ks + hi * 16);
    v8bf a01 = *(const v8bf*)(arow0 + ks + hi * 16 + 8);
    v8bf a10 = *(const v8bf*)(arow1 + ks + hi * 16);
    v8bf a11 = *(const v8bf*)(arow1 + ks + hi * 16 + 8);
#pragma unroll
    for (int t = 0; t < 8; t++) {
      A0[t] = a00[t]; A0[t + 8] = a01[t];
      A1[t] = a10[t]; A1[t + 8] = a11[t];
    }
#pragma unroll
    for (int nt = 0; nt < 4; nt++) {
      const __bf16* bp = wkT + (size_t)(nBase + nt * 16 + ln) * DEM + ks + hi * 16;
      v16bf Bm = *(const v16bf*)bp;          // B: lane=col, 16 halfs along K
      acc[0][nt] = __builtin_amdgcn_wmma_f32_16x16x32_bf16(
          false, A0, false, Bm, (short)0, acc[0][nt], false, false);
      acc[1][nt] = __builtin_amdgcn_wmma_f32_16x16x32_bf16(
          false, A1, false, Bm, (short)0, acc[1][nt], false, false);
    }
  }
#pragma unroll
  for (int mt = 0; mt < 2; mt++) {
#pragma unroll
    for (int nt = 0; nt < 4; nt++) {
      int col = nBase + nt * 16 + ln;
      float bv = bias[col];
      int which = col >> 10;          // 0=Q 1=K 2=V
      int cm = col & 1023;
      int h = cm >> 6, d = cm & 63;
#pragma unroll
      for (int r = 0; r < 8; r++) {
        int m = mBase + mt * 16 + r + 8 * hi;   // D layout: M = r + 8*hi
        int b = m >> 11, s = m & 2047;
        float v = acc[mt][nt][r] + bv;
        size_t bh = (size_t)(b * NH + h);
        if (which == 0)      qb  [(bh * SEQ + s) * DKK + d] = f2bf(v * 0.125f);
        else if (which == 1) kbuf[(bh * SEQ + s) * DKK + d] = f2bf(v);
        else                 vT  [(bh * DKK + d) * SEQ + s] = f2bf(v);
      }
    }
  }
}

// ---------------------------------------------------------------------------
// 3) Fused attention: per workgroup = one (b,h) and 16 query rows.
//    scores (WMMA) -> LDS (16x2048 f32 = 128KB, fits CDNA5's 320KB WGP LDS)
//    -> mask -> softmax (shfl_xor reductions) -> fp32 probs to d_out
//    -> fused attn@V (WMMA, probs re-read from LDS as bf16) -> ds_add_f32
//    cross-wave combine -> bf16 blended tile.
// ---------------------------------------------------------------------------
__global__ __launch_bounds__(256) void k_attn(
    const __bf16* __restrict__ qb, const __bf16* __restrict__ kbuf,
    const __bf16* __restrict__ vT, const float* __restrict__ mask,
    float* __restrict__ attn_out, __bf16* __restrict__ blend) {
  extern __shared__ float lds[];
  float* sc = lds;                    // [16][2048] scores/probs
  float* bl = lds + 16 * 2048;        // [16][64] blended accumulator

  const int w = threadIdx.x >> 5, lane = threadIdx.x & 31;
  const int hi = lane >> 4, ln = lane & 15;
  const int bh = blockIdx.x >> 7;     // b*16+h
  const int qBase = (blockIdx.x & 127) * 16;
  const int b = bh >> 4;

  // Q fragment (16x64 -> two 16x32 A-fragments)
  const __bf16* qrow = qb + ((size_t)bh * SEQ + qBase + ln) * DKK;
  v16bf A0, A1;
  {
    v8bf a0 = *(const v8bf*)(qrow + hi * 16);
    v8bf a1 = *(const v8bf*)(qrow + hi * 16 + 8);
    v8bf a2 = *(const v8bf*)(qrow + 32 + hi * 16);
    v8bf a3 = *(const v8bf*)(qrow + 32 + hi * 16 + 8);
#pragma unroll
    for (int t = 0; t < 8; t++) {
      A0[t] = a0[t]; A0[t + 8] = a1[t];
      A1[t] = a2[t]; A1[t + 8] = a3[t];
    }
  }
  // scores: wave w owns key columns [w*256, w*256+256)
  for (int nt = 0; nt < 16; nt++) {
    int nBase = w * 256 + nt * 16;
    const __bf16* krow = kbuf + ((size_t)bh * SEQ + nBase + ln) * DKK;
    v16bf B0 = *(const v16bf*)(krow + hi * 16);          // B = K^T, contiguous d
    v16bf B1 = *(const v16bf*)(krow + 32 + hi * 16);
    v8f acc = {};
    acc = __builtin_amdgcn_wmma_f32_16x16x32_bf16(false, A0, false, B0, (short)0, acc, false, false);
    acc = __builtin_amdgcn_wmma_f32_16x16x32_bf16(false, A1, false, B1, (short)0, acc, false, false);
    int col = nBase + ln;
    float am = 1.0f - mask[b * SEQ + col];
    if (am == 1.0f) am = NEG_BIG;
#pragma unroll
    for (int r = 0; r < 8; r++) sc[(r + 8 * hi) * 2048 + col] = acc[r] + am;
  }
  __syncthreads();

  // softmax: wave w handles rows 2w, 2w+1
#pragma unroll
  for (int rr = 0; rr < 2; rr++) {
    int row = w * 2 + rr;
    float* srow = sc + row * 2048;
    float mx = -3.4e38f;
    for (int c = lane; c < 2048; c += 32) mx = fmaxf(mx, srow[c]);
#pragma unroll
    for (int o = 16; o > 0; o >>= 1) mx = fmaxf(mx, __shfl_xor(mx, o, 32));
    float sum = 0.f;
    for (int c = lane; c < 2048; c += 32) {
      float e = __expf(srow[c] - mx);
      srow[c] = e; sum += e;
    }
#pragma unroll
    for (int o = 16; o > 0; o >>= 1) sum += __shfl_xor(sum, o, 32);
    float inv = 1.0f / sum;
    float* arow_out = attn_out + ((size_t)bh * SEQ + qBase + row) * SEQ;
    for (int c = lane; c < 2048; c += 32) {
      float p = srow[c] * inv;
      srow[c] = p;
      arow_out[c] = p;          // fp32 attention probs (2nd output)
    }
  }
  for (int i = threadIdx.x; i < 16 * 64; i += 256) bl[i] = 0.f;
  __syncthreads();

  // attn @ V: wave w -> d-tile (w&3), K-half (w>>2)*1024
  {
    int nt = w & 3, kh = w >> 2;
    const __bf16* vbase = vT + (size_t)bh * DKK * SEQ;   // [64][2048]
    v8f acc = {};
    for (int ks = 0; ks < 1024; ks += 32) {
      int kk = kh * 1024 + ks;
      v16bf A;
      const float* sp = sc + ln * 2048 + kk + 16 * hi;
#pragma unroll
      for (int t = 0; t < 8; t++) { A[t] = f2bf(sp[t]); A[t + 8] = f2bf(sp[t + 8]); }
      v16bf Bm = *(const v16bf*)(vbase + (size_t)(nt * 16 + ln) * SEQ + kk + hi * 16);
      acc = __builtin_amdgcn_wmma_f32_16x16x32_bf16(false, A, false, Bm, (short)0, acc, false, false);
    }
#pragma unroll
    for (int r = 0; r < 8; r++)
      atomicAdd(&bl[(r + 8 * hi) * 64 + nt * 16 + ln], acc[r]);   // ds_add_f32
  }
  __syncthreads();

  // merged-head blended tile -> bf16 [b*2048+s][1024] at col h*64+d
  int h = bh & 15;
  for (int i = threadIdx.x; i < 16 * 64; i += 256) {
    int m = i >> 6, d = i & 63;
    blend[((size_t)b * SEQ + qBase + m) * DEM + h * 64 + d] = f2bf(bl[i]);
  }
}

// ---------------------------------------------------------------------------
// 4) output projection: [4096 x 1024] x [1024 x 1024] + bias -> fp32 out
//    Same 2x4 register blocking as k_qkv.
// ---------------------------------------------------------------------------
__global__ __launch_bounds__(256) void k_out(
    const __bf16* __restrict__ xb, const __bf16* __restrict__ wT,
    const float* __restrict__ bias, float* __restrict__ out) {
  const int w = threadIdx.x >> 5, lane = threadIdx.x & 31;
  const int hi = lane >> 4, ln = lane & 15;
  const int mBase = blockIdx.x * 64 + (w & 1) * 32;
  const int nBase = blockIdx.y * 256 + (w >> 1) * 64;

  v8f acc[2][4] = {};
  const __bf16* arow0 = xb + (size_t)(mBase + ln) * DEM;
  const __bf16* arow1 = xb + (size_t)(mBase + 16 + ln) * DEM;
  for (int ks = 0; ks < DEM; ks += 32) {
    v16bf A0, A1;
    v8bf a00 = *(const v8bf*)(arow0 + ks + hi * 16);
    v8bf a01 = *(const v8bf*)(arow0 + ks + hi * 16 + 8);
    v8bf a10 = *(const v8bf*)(arow1 + ks + hi * 16);
    v8bf a11 = *(const v8bf*)(arow1 + ks + hi * 16 + 8);
#pragma unroll
    for (int t = 0; t < 8; t++) {
      A0[t] = a00[t]; A0[t + 8] = a01[t];
      A1[t] = a10[t]; A1[t + 8] = a11[t];
    }
#pragma unroll
    for (int nt = 0; nt < 4; nt++) {
      v16bf Bm = *(const v16bf*)(wT + (size_t)(nBase + nt * 16 + ln) * DEM + ks + hi * 16);
      acc[0][nt] = __builtin_amdgcn_wmma_f32_16x16x32_bf16(
          false, A0, false, Bm, (short)0, acc[0][nt], false, false);
      acc[1][nt] = __builtin_amdgcn_wmma_f32_16x16x32_bf16(
          false, A1, false, Bm, (short)0, acc[1][nt], false, false);
    }
  }
#pragma unroll
  for (int mt = 0; mt < 2; mt++) {
#pragma unroll
    for (int nt = 0; nt < 4; nt++) {
      int col = nBase + nt * 16 + ln;
      float bv = bias[col];
#pragma unroll
      for (int r = 0; r < 8; r++) {
        int m = mBase + mt * 16 + r + 8 * hi;
        out[(size_t)m * DEM + col] = acc[mt][nt][r] + bv;
      }
    }
  }
}

// ---------------------------------------------------------------------------
extern "C" void kernel_launch(void* const* d_in, const int* in_sizes, int n_in,
                              void* d_out, int out_size, void* d_ws, size_t ws_size,
                              hipStream_t stream) {
  const float* x    = (const float*)d_in[0];
  const float* mask = (const float*)d_in[1];
  const float* wk_w = (const float*)d_in[2];
  const float* wk_b = (const float*)d_in[3];
  const float* wo_w = (const float*)d_in[4];
  const float* wo_b = (const float*)d_in[5];

  char* ws = (char*)d_ws;
  __bf16* xb    = (__bf16*)(ws);                 //  8 MB  x bf16
  __bf16* wkT   = (__bf16*)(ws + 8388608);       //  6 MB  wk^T bf16
  __bf16* woT   = (__bf16*)(ws + 14680064);      //  2 MB  wo^T bf16
  __bf16* qb    = (__bf16*)(ws + 16777216);      //  8 MB  Q [b,h,s,dk]
  __bf16* kb    = (__bf16*)(ws + 25165824);      //  8 MB  K [b,h,s,dk]
  __bf16* vT    = (__bf16*)(ws + 33554432);      //  8 MB  V [b,h,dk,s]
  __bf16* blend = (__bf16*)(ws + 41943040);      //  8 MB  blended bf16
                                                 // total 48 MB

  float* out  = (float*)d_out;
  float* attn = out + (size_t)BATCH * SEQ * DEM; // outputs concatenated: out, attn

  k_convert<<<16384, 256, 0, stream>>>(x, wk_w, wo_w, xb, wkT, woT);
  k_qkv<<<dim3(64, 12), 256, 0, stream>>>(xb, wkT, wk_b, qb, kb, vT);
  const int ldsBytes = (16 * 2048 + 16 * 64) * 4;   // 132 KB (CDNA5 WGP: 320 KB)
  k_attn<<<4096, 256, ldsBytes, stream>>>(qb, kb, vT, mask, attn, blend);
  k_out<<<dim3(64, 4), 256, 0, stream>>>(blend, woT, wo_b, out);
}
